// AdaConv_33062658244826
// MI455X (gfx1250) — compile-verified, matrix operand order checked by reading, not run on hardware
//
#include <hip/hip_runtime.h>
#include <hip/hip_bf16.h>

typedef __attribute__((ext_vector_type(16))) _Float16 v16h;
typedef __attribute__((ext_vector_type(8)))  _Float16 v8h;
typedef __attribute__((ext_vector_type(8)))  float    v8f;
typedef int v4i __attribute__((vector_size(16)));

#define BATCH 8
#define HH 64
#define WW 64
#define C_IN 256
#define C_OUT 1024
#define HW (HH*WW)              // 4096
#define BHW (BATCH*HW)          // 32768
#define KDIM (9*C_IN)           // 2304

// GEMM tiling
#define BM 128
#define BN 128
#define BK 32
#define LDPAD 40                // f16 elements per LDS row (80B pitch, conflict-free)

#ifndef __has_builtin
#define __has_builtin(x) 0
#endif
#if __has_builtin(__builtin_amdgcn_global_load_async_to_lds_b128)
#define HAVE_ASYNC 1
#else
#define HAVE_ASYNC 0
#endif

// async-to-LDS builtin wants v4i pointers in explicit address spaces:
//   param0: global (AS1) int4*, param1: LDS (AS3) int4*
#define ASG(p) ((__attribute__((address_space(1))) v4i*)(p))
#define ASL(p) ((__attribute__((address_space(3))) v4i*)(p))

// ---------------------------------------------------------------------------
// Kernel 1: per-(b,c) instance-norm statistics: mean and 1/(std+eps)
// ---------------------------------------------------------------------------
__global__ void __launch_bounds__(256) k_stats(const float* __restrict__ x,
                                               float* __restrict__ mean,
                                               float* __restrict__ rinv) {
    const int bc = blockIdx.x;              // 0..B*C-1
    const int b  = bc >> 8;
    const int c  = bc & (C_IN - 1);
    const float* px = x + (size_t)b * HW * C_IN + c;

    float s = 0.f, s2 = 0.f;
    for (int p = threadIdx.x; p < HW; p += 256) {
        float v = px[(size_t)p * C_IN];
        s += v; s2 += v * v;
    }
    __shared__ float sh[512];
    sh[threadIdx.x] = s;
    sh[256 + threadIdx.x] = s2;
    __syncthreads();
    for (int o = 128; o > 0; o >>= 1) {
        if (threadIdx.x < o) {
            sh[threadIdx.x]       += sh[threadIdx.x + o];
            sh[256 + threadIdx.x] += sh[256 + threadIdx.x + o];
        }
        __syncthreads();
    }
    if (threadIdx.x == 0) {
        float m   = sh[0] * (1.0f / HW);
        float var = sh[256] * (1.0f / HW) - m * m;
        var = var < 0.f ? 0.f : var;
        mean[bc] = m;
        rinv[bc] = 1.0f / (sqrtf(var) + 1e-5f);
    }
}

// ---------------------------------------------------------------------------
// Kernel 2: fused instance-norm + per-sample depthwise 3x3 + pointwise + bias
//           -> f16 activations [B,H,W,C]
// ---------------------------------------------------------------------------
__global__ void __launch_bounds__(256) k_prep(const float* __restrict__ x,
                                              const float* __restrict__ wsp,   // [B,3,3,1,C]
                                              const float* __restrict__ wpw,   // [B,1,1,1,C]
                                              const float* __restrict__ bias,  // [B,C]
                                              const float* __restrict__ mean,
                                              const float* __restrict__ rinv,
                                              _Float16* __restrict__ act) {
    const int idx = blockIdx.x * 256 + threadIdx.x;   // over B*H*W*C
    const int c = idx & (C_IN - 1);
    int t = idx >> 8;
    const int w = t & (WW - 1); t >>= 6;
    const int h = t & (HH - 1);
    const int b = t >> 6;
    const int bc = (b << 8) | c;

    const float m  = mean[bc];
    const float iv = rinv[bc];
    const float* wsb = wsp + (size_t)b * 9 * C_IN + c;
    const float* xb  = x   + (size_t)b * HW * C_IN + c;

    float acc = 0.f;
#pragma unroll
    for (int ky = 0; ky < 3; ++ky) {
        const int hh = h + ky - 1;
        if (hh < 0 || hh >= HH) continue;
#pragma unroll
        for (int kx = 0; kx < 3; ++kx) {
            const int ww2 = w + kx - 1;
            if (ww2 < 0 || ww2 >= WW) continue;
            float xv = xb[(size_t)(hh * WW + ww2) * C_IN];
            acc += wsb[(ky * 3 + kx) * C_IN] * ((xv - m) * iv);
        }
    }
    float y = acc * wpw[bc] + bias[bc];
    act[idx] = (_Float16)y;
}

// ---------------------------------------------------------------------------
// Kernel 3: conv_w [3,3,C,OUT] fp32 -> Wt[n][k] f16 (k = tap*256 + c)
// ---------------------------------------------------------------------------
__global__ void __launch_bounds__(256) k_wt(const float* __restrict__ cw,
                                            _Float16* __restrict__ wt) {
    const int id = blockIdx.x * 256 + threadIdx.x;    // over OUT*KDIM
    const int n = id / KDIM;
    const int k = id - n * KDIM;
    wt[id] = (_Float16)cw[(size_t)k * C_OUT + n];
}

// ---------------------------------------------------------------------------
// WMMA A/B fragment load per the 16-bit A layout:
// lane = row/col (lane&15); frag[0..7]=K half*8.., frag[8..15]=K 16+half*8..
// ---------------------------------------------------------------------------
static __device__ inline v16h load_frag(const _Float16* p) {
    v8h lo = *(const v8h*)(p);
    v8h hi = *(const v8h*)(p + 16);
    v16h r;
#pragma unroll
    for (int i = 0; i < 8; ++i) { r[i] = lo[i]; r[i + 8] = hi[i]; }
    return r;
}

// ---------------------------------------------------------------------------
// Kernel 4: implicit-GEMM 3x3 conv via v_wmma_f32_16x16x32_f16
//           M = B*H*W = 32768, N = OUT = 1024, K = 9*C = 2304
//           Block tile 128x128, wave tile 32x64 (8 WMMA / K-step / wave)
// ---------------------------------------------------------------------------
__global__ void __launch_bounds__(256) k_gemm(const _Float16* __restrict__ act,
                                              const _Float16* __restrict__ wt,
                                              const float* __restrict__ convb,
                                              float* __restrict__ out) {
    __shared__ __align__(16) _Float16 shA[BM * LDPAD];   // 10240 B
    __shared__ __align__(16) _Float16 shB[BN * LDPAD];   // 10240 B

    const int tid   = threadIdx.x;
    const int tileM = blockIdx.x * BM;   // 256 tiles
    const int tileN = blockIdx.y * BN;   // 8 tiles

    const int lane  = tid & 31;
    const int wv    = tid >> 5;          // 8 waves: 4 in M x 2 in N
    const int waveM = (wv & 3) * 32;
    const int waveN = (wv >> 2) * 64;
    const int lrow  = lane & 15;
    const int half  = lane >> 4;

    v8f acc[2][4];
#pragma unroll
    for (int mi = 0; mi < 2; ++mi)
#pragma unroll
        for (int ni = 0; ni < 4; ++ni)
            acc[mi][ni] = (v8f){};

    for (int ks = 0; ks < KDIM / BK; ++ks) {          // 72 K-steps
        const int tap = ks >> 3;                       // 3x3 tap (256/32 = 8 steps per tap)
        const int ky  = tap / 3;
        const int kx  = tap - ky * 3;
        const int c0  = (ks & 7) * BK;                 // channel base within tap

        // ---- A tile (im2col on the fly): 128 rows x 32 k, zero-fill OOB ----
#pragma unroll
        for (int i = 0; i < 2; ++i) {
            const int id  = tid + i * 256;             // 512 chunks of 8 f16
            const int row = id >> 2;
            const int kc  = (id & 3) * 8;
            const int m   = tileM + row;
            const int b   = m >> 12;
            const int hw  = m & (HW - 1);
            const int h   = (hw >> 6) + ky - 1;
            const int w   = (hw & 63) + kx - 1;
            uint4 val = make_uint4(0u, 0u, 0u, 0u);
            if (h >= 0 && h < HH && w >= 0 && w < WW) {
                val = *(const uint4*)(act +
                        ((size_t)b * HW + h * WW + w) * C_IN + c0 + kc);
            }
            *(uint4*)(&shA[row * LDPAD + kc]) = val;
        }
        // ---- B tile: 128 cols x 32 k, always in-bounds -> async direct to LDS ----
#pragma unroll
        for (int i = 0; i < 2; ++i) {
            const int id  = tid + i * 256;
            const int col = id >> 2;
            const int kc  = (id & 3) * 8;
            const _Float16* gp = wt + (size_t)(tileN + col) * KDIM + ks * BK + kc;
            _Float16* lp = &shB[col * LDPAD + kc];
#if HAVE_ASYNC
            __builtin_amdgcn_global_load_async_to_lds_b128(ASG(gp), ASL(lp), 0, 0);
#else
            *(uint4*)lp = *(const uint4*)gp;
#endif
        }
#if HAVE_ASYNC
#if __has_builtin(__builtin_amdgcn_s_wait_asynccnt)
        __builtin_amdgcn_s_wait_asynccnt(0);
#else
        asm volatile("s_wait_asynccnt 0x0" ::: "memory");
#endif
#endif
        __syncthreads();

        const v16h a0 = load_frag(&shA[(waveM      + lrow) * LDPAD + half * 8]);
        const v16h a1 = load_frag(&shA[(waveM + 16 + lrow) * LDPAD + half * 8]);
        v16h bf[4];
#pragma unroll
        for (int ni = 0; ni < 4; ++ni)
            bf[ni] = load_frag(&shB[(waveN + ni * 16 + lrow) * LDPAD + half * 8]);

#pragma unroll
        for (int ni = 0; ni < 4; ++ni) {
            acc[0][ni] = __builtin_amdgcn_wmma_f32_16x16x32_f16(false, a0, false, bf[ni],
                                                                (short)0, acc[0][ni], false, false);
            acc[1][ni] = __builtin_amdgcn_wmma_f32_16x16x32_f16(false, a1, false, bf[ni],
                                                                (short)0, acc[1][ni], false, false);
        }
        __syncthreads();
    }

    // ---- epilogue: C/D layout row = v + 8*half, col = lane&15; add conv_b ----
#pragma unroll
    for (int mi = 0; mi < 2; ++mi) {
        const int row = tileM + waveM + mi * 16 + half * 8;
#pragma unroll
        for (int ni = 0; ni < 4; ++ni) {
            const int col = tileN + waveN + ni * 16 + lrow;
            const float bn = convb[col];
#pragma unroll
            for (int v = 0; v < 8; ++v) {
                out[(size_t)(row + v) * C_OUT + col] = acc[mi][ni][v] + bn;
            }
        }
    }
}

// ---------------------------------------------------------------------------
// Launcher
// ---------------------------------------------------------------------------
extern "C" void kernel_launch(void* const* d_in, const int* in_sizes, int n_in,
                              void* d_out, int out_size, void* d_ws, size_t ws_size,
                              hipStream_t stream) {
    (void)in_sizes; (void)n_in; (void)out_size; (void)ws_size;

    const float* x      = (const float*)d_in[0];  // [8,64,64,256]
    const float* wsp    = (const float*)d_in[1];  // [8,3,3,1,256]
    const float* wpw    = (const float*)d_in[2];  // [8,1,1,1,256]
    const float* bias   = (const float*)d_in[3];  // [8,256]
    const float* conv_w = (const float*)d_in[4];  // [3,3,256,1024]
    const float* conv_b = (const float*)d_in[5];  // [1024]
    float* out = (float*)d_out;                   // [8,64,64,1024]

    char* ws = (char*)d_ws;
    const size_t WT_BYTES  = (size_t)C_OUT * KDIM * sizeof(_Float16);      // 4,718,592
    const size_t ACT_BYTES = (size_t)BHW * C_IN * sizeof(_Float16);        // 16,777,216
    _Float16* wt   = (_Float16*)ws;
    _Float16* act  = (_Float16*)(ws + WT_BYTES);
    float*    mean = (float*)(ws + WT_BYTES + ACT_BYTES);
    float*    rinv = mean + BATCH * C_IN;

    // 1) weight convert/transpose -> f16 K-contiguous
    k_wt<<<(C_OUT * KDIM) / 256, 256, 0, stream>>>(conv_w, wt);
    // 2) instance-norm stats
    k_stats<<<BATCH * C_IN, 256, 0, stream>>>(x, mean, rinv);
    // 3) fused norm + depthwise + pointwise + bias -> f16 activations
    k_prep<<<(BHW * C_IN) / 256, 256, 0, stream>>>(x, wsp, wpw, bias, mean, rinv, act);
    // 4) WMMA implicit GEMM (32768 x 1024 x 2304), block 128x128
    dim3 grid(BHW / BM, C_OUT / BN);
    k_gemm<<<grid, 256, 0, stream>>>(act, wt, conv_b, out);
}
